// DimeNetInteraction_695784702036
// MI455X (gfx1250) — matrix operand order repaired and checked by reading, not compiled, not run on hardware
//
#include <hip/hip_runtime.h>
#include <hip/hip_bf16.h>

// ---------------------------------------------------------------------------
// Types for CDNA5 WMMA (wave32, 16x16x32 bf16 -> f32) and TDM descriptors
// ---------------------------------------------------------------------------
typedef __attribute__((ext_vector_type(16))) __bf16 v16bf;
typedef __attribute__((ext_vector_type(8)))  __bf16 v8bf;
typedef __attribute__((ext_vector_type(8)))  float  v8f;
typedef __attribute__((ext_vector_type(4)))  unsigned int u32x4;
typedef __attribute__((ext_vector_type(8)))  int          i32x8;
typedef __attribute__((ext_vector_type(4)))  int          i32x4;

#if defined(__has_builtin)
#  if __has_builtin(__builtin_amdgcn_tensor_load_to_lds)
#    define HAVE_TDM 1
#  endif
#endif
#ifndef HAVE_TDM
#  define HAVE_TDM 0
#endif

#if HAVE_TDM
#  if __has_include(<hip/amd_detail/amd_gfx1250_TDM.h>)
     // therock/clang-23 lane: 6-arg form (extra i32x8 group before cpol)
#    define TDM_LOAD(g0, g1, g2, g3) \
        __builtin_amdgcn_tensor_load_to_lds((g0), (g1), (g2), (g3), (i32x8)(0), 0)
#  else
     // ROCm 7.2 / clang-22 lane: 5-arg form
#    define TDM_LOAD(g0, g1, g2, g3) \
        __builtin_amdgcn_tensor_load_to_lds((g0), (g1), (g2), (g3), 0)
#  endif
#endif

__device__ __forceinline__ void wait_tensorcnt0() {
#if defined(__has_builtin) && __has_builtin(__builtin_amdgcn_s_wait_tensorcnt)
    __builtin_amdgcn_s_wait_tensorcnt(0);
#else
    asm volatile("s_wait_tensorcnt 0x0" ::: "memory");
#endif
}

__device__ __forceinline__ unsigned short f32_to_bf16(float f) {
    unsigned int u = __builtin_bit_cast(unsigned int, f);
    u += 0x7fffu + ((u >> 16) & 1u);   // round-to-nearest-even
    return (unsigned short)(u >> 16);
}

// ---------------------------------------------------------------------------
// Weight convert + transpose:  W fp32 [K][256] row-major  ->  Wt bf16 [256][K]
// ---------------------------------------------------------------------------
__global__ void k_convert_transpose(const float* __restrict__ W,
                                    unsigned short* __restrict__ Wt, int K) {
    int idx = blockIdx.x * 256 + threadIdx.x;        // total = K*256
    if (idx >= K * 256) return;
    int k = idx >> 8;
    int n = idx & 255;
    Wt[n * K + k] = f32_to_bf16(W[idx]);
}

__global__ void k_zero_f32(float* __restrict__ p, long long n4) {
    long long i = (long long)blockIdx.x * blockDim.x + threadIdx.x;
    if (i < n4) ((float4*)p)[i] = make_float4(0.f, 0.f, 0.f, 0.f);
}

// agg[dst[e]] += msg[src[e]] * elg[e]
__global__ void k_lg_scatter(const float* __restrict__ msg,
                             const float* __restrict__ elg,
                             const int* __restrict__ src,
                             const int* __restrict__ dst,
                             float* __restrict__ agg, int nEdges) {
    long long t = (long long)blockIdx.x * blockDim.x + threadIdx.x;
    long long total = (long long)nEdges * 64;
    if (t >= total) return;
    int e = (int)(t >> 6);
    int c = (int)(t & 63) << 2;
    int s = src[e], d = dst[e];
    float4 mv = *(const float4*)(msg + (long long)s * 256 + c);
    float4 ev = *(const float4*)(elg + (long long)e * 256 + c);
    float* o  = agg + (long long)d * 256 + c;
    atomicAdd(o + 0, mv.x * ev.x);
    atomicAdd(o + 1, mv.y * ev.y);
    atomicAdd(o + 2, mv.z * ev.z);
    atomicAdd(o + 3, mv.w * ev.w);
}

// node[dst[e]] += ee[e] * x4[e]
__global__ void k_node_scatter(const float* __restrict__ ee,
                               const float* __restrict__ x4,
                               const int* __restrict__ dst,
                               float* __restrict__ node, int nEdges) {
    long long t = (long long)blockIdx.x * blockDim.x + threadIdx.x;
    long long total = (long long)nEdges * 64;
    if (t >= total) return;
    int e = (int)(t >> 6);
    int c = (int)(t & 63) << 2;
    int d = dst[e];
    float4 a = *(const float4*)(ee + (long long)e * 256 + c);
    float4 b = *(const float4*)(x4 + (long long)e * 256 + c);
    float* o = node + (long long)d * 256 + c;
    atomicAdd(o + 0, a.x * b.x);
    atomicAdd(o + 1, a.y * b.y);
    atomicAdd(o + 2, a.z * b.z);
    atomicAdd(o + 3, a.w * b.w);
}

// ---------------------------------------------------------------------------
// WMMA GEMM:  out[M,256] = epilogue( X[M,K] @ W[K,256] )
//   Wt panel (bf16, pre-transposed [256][K]) -> LDS via TDM (1D tile, DMA
//     overlaps the cooperative X staging below)
//   X tile   (fp32 -> bf16 convert while staging) -> LDS [64][K]
// 256 threads = 8 wave32 (4 M-strips x 2 N-groups); 64x256 output tile.
// LDS layout: [0, 256*K*2)  = Wt bf16 panel
//             [256*K*2, ..) = X bf16 tile [64][K]
// K is compile-time: K-loop fully unrolls -> immediate LDS offsets, register
// rotation across B fragments (avoids WMMA WAR hazard nops).
// ---------------------------------------------------------------------------
template <int K>
__global__ void __launch_bounds__(256)
k_gemm_wmma(const float* __restrict__ X,
            const unsigned short* __restrict__ Wt,
            const float* __restrict__ bias,
            const float* __restrict__ res,
            float* __restrict__ out,
            int M, int do_relu) {
    extern __shared__ unsigned short smem[];
    unsigned short* Wl = smem;                       // [256][K] bf16 bits
    unsigned short* Xl = smem + 256 * K;             // [64][K]  bf16 bits

    const int tid   = threadIdx.x;
    const int mbase = blockIdx.x * 64;

#if HAVE_TDM
    // ---- weight panel via Tensor Data Mover (1D tile of 8-byte elems) ----
    if (tid == 0) {
        u32x4 g0; i32x8 g1; i32x4 gz;
        gz[0] = gz[1] = gz[2] = gz[3] = 0;
        unsigned n8 = ((unsigned)(256 * K)) >> 2;           // 8B elems (bf16 x4)
        unsigned long long ga = (unsigned long long)(uintptr_t)Wt;
        g0[0] = 1u;                                         // count=1, user mode
        g0[1] = 0u;                                         // lds_addr = 0
        g0[2] = (unsigned)ga;                               // global_addr[31:0]
        g0[3] = (unsigned)((ga >> 32) & 0x1ffffffu) | (2u << 30); // addr[56:32]|type=2
        g1[0] = (int)(3u << 16);                            // data_size=8B, mask=0
        g1[1] = (int)((n8 & 0xffffu) << 16);                // tensor_dim0[15:0]
        g1[2] = (int)(((n8 >> 16) & 0xffffu) | (1u << 16)); // td0 hi | tensor_dim1=1
        g1[3] = (int)(n8 << 16);                            // td1 hi(0) | tile_dim0
        g1[4] = 0;                                          // tile_dim1/2 unused
        g1[5] = (int)n8;                                    // tensor_dim0_stride lo
        g1[6] = 0;
        g1[7] = 0;
        TDM_LOAD(g0, g1, gz, gz);
    }
#else
    {
        const uint4* s = (const uint4*)Wt;
        uint4*       d = (uint4*)Wl;
        const int cnt = (256 * K) >> 3;
        for (int i = tid; i < cnt; i += 256) d[i] = s[i];
    }
#endif
    // ---- X tile: fp32 -> bf16 convert while staging (overlaps the DMA) ----
    {
        const int Kq  = K >> 2;
        const int cnt = 64 * Kq;
        for (int i = tid; i < cnt; i += 256) {
            int row  = i / Kq;
            int col  = (i - row * Kq) << 2;
            int grow = mbase + row;
            float4 v = (grow < M) ? *(const float4*)(X + (long long)grow * K + col)
                                  : make_float4(0.f, 0.f, 0.f, 0.f);
            unsigned long long pk =
                  (unsigned long long)f32_to_bf16(v.x)
                | ((unsigned long long)f32_to_bf16(v.y) << 16)
                | ((unsigned long long)f32_to_bf16(v.z) << 32)
                | ((unsigned long long)f32_to_bf16(v.w) << 48);
            *(unsigned long long*)(Xl + row * K + col) = pk;
        }
    }
#if HAVE_TDM
    wait_tensorcnt0();
#endif
    __syncthreads();

    const int wave = tid >> 5, lane = tid & 31;
    const int wm   = wave & 3;          // M strip (16 rows)
    const int wn   = wave >> 2;         // N group (128 cols)
    const int half = lane >> 4;
    const int lrow = lane & 15;

    v8f acc[8];
#pragma unroll
    for (int t = 0; t < 8; ++t)
#pragma unroll
        for (int j = 0; j < 8; ++j) acc[t][j] = 0.0f;

    // A fragment base (16-bit A 16x32 layout: K = half*8 + 0..7 and +16)
    const unsigned short* Arow = Xl + (wm * 16 + lrow) * K + half * 8;
    const unsigned short* Bbase = Wl + (wn * 128 + lrow) * K + half * 8;

#pragma unroll
    for (int kk = 0; kk < K; kk += 32) {
        v16bf a;
        {
            v8bf alo = *(const v8bf*)(Arow + kk);
            v8bf ahi = *(const v8bf*)(Arow + kk + 16);
#pragma unroll
            for (int e = 0; e < 8; ++e) { a[e] = alo[e]; a[e + 8] = ahi[e]; }
        }
#pragma unroll
        for (int t = 0; t < 8; ++t) {
            // B 32x16: lane = column, K = half*8 + 0..7 and +16 (same as A)
            const unsigned short* Bp = Bbase + t * 16 * K + kk;
            v16bf b;
            v8bf blo = *(const v8bf*)(Bp);
            v8bf bhi = *(const v8bf*)(Bp + 16);
#pragma unroll
            for (int e = 0; e < 8; ++e) { b[e] = blo[e]; b[e + 8] = bhi[e]; }
            acc[t] = __builtin_amdgcn_wmma_f32_16x16x32_bf16(
                false, a, false, b, (short)0, acc[t], false, false);
        }
    }

    // ---- epilogue: bias -> relu -> residual-add -> store ----
#pragma unroll
    for (int t = 0; t < 8; ++t) {
        int col = wn * 128 + t * 16 + lrow;
#pragma unroll
        for (int i = 0; i < 8; ++i) {
            int row = mbase + wm * 16 + half * 8 + i;
            if (row < M) {
                float v = acc[t][i];
                if (bias)    v += bias[col];
                if (do_relu) v  = fmaxf(v, 0.0f);
                if (res)     v += res[(long long)row * 256 + col];
                out[(long long)row * 256 + col] = v;
            }
        }
    }
}

// ---------------------------------------------------------------------------
// Host-side helpers
// ---------------------------------------------------------------------------
static inline void run_gemm(const float* X, const unsigned short* Wt,
                            const float* bias, const float* res, float* out,
                            int M, int K, int relu, hipStream_t s) {
    size_t smem = (size_t)640 * (size_t)K;   // 256*K*2 (W) + 64*K*2 (X), bf16
    dim3 grid((M + 63) / 64);
    if (K == 64) {
        hipFuncSetAttribute(reinterpret_cast<const void*>(k_gemm_wmma<64>),
                            hipFuncAttributeMaxDynamicSharedMemorySize, (int)smem);
        k_gemm_wmma<64><<<grid, 256, smem, s>>>(X, Wt, bias, res, out, M, relu);
    } else {
        hipFuncSetAttribute(reinterpret_cast<const void*>(k_gemm_wmma<256>),
                            hipFuncAttributeMaxDynamicSharedMemorySize, (int)smem);
        k_gemm_wmma<256><<<grid, 256, smem, s>>>(X, Wt, bias, res, out, M, relu);
    }
}

extern "C" void kernel_launch(void* const* d_in, const int* in_sizes, int n_in,
                              void* d_out, int out_size, void* d_ws, size_t ws_size,
                              hipStream_t stream) {
    const float* msg    = (const float*)d_in[0];
    const float* gattr  = (const float*)d_in[1];
    const int*   gidx   = (const int*)  d_in[2];   // [2,E]
    const float* lgattr = (const float*)d_in[3];
    const int*   lgidx  = (const int*)  d_in[4];   // [2,ELG]
    const float* W1   = (const float*)d_in[6],  *b1   = (const float*)d_in[7];
    const float* Wlge = (const float*)d_in[8];
    const float* Wlgm = (const float*)d_in[9];
    const float* Wr1a = (const float*)d_in[10], *br1a = (const float*)d_in[11];
    const float* Wr1b = (const float*)d_in[12], *br1b = (const float*)d_in[13];
    const float* Wr2a = (const float*)d_in[14], *br2a = (const float*)d_in[15];
    const float* Wr2b = (const float*)d_in[16], *br2b = (const float*)d_in[17];
    const float* Wr3a = (const float*)d_in[18], *br3a = (const float*)d_in[19];
    const float* Wr3b = (const float*)d_in[20], *br3b = (const float*)d_in[21];
    const float* W2   = (const float*)d_in[22], *b2   = (const float*)d_in[23];
    const float* Woe  = (const float*)d_in[24], *boe  = (const float*)d_in[25];
    const float* Wol0 = (const float*)d_in[26], *bol0 = (const float*)d_in[27];
    const float* Wo4  = (const float*)d_in[28];

    const int D   = 256;
    const int E   = in_sizes[0] / D;
    const int ELG = in_sizes[4] / 2;
    const int N   = out_size / D - E;
    const long long ED = (long long)E * D;
    const long long ND = (long long)N * D;

    // workspace: 5 E-sized f32 buffers (B1..B3 alias the ELG*D tensor) + bf16 weights
    float* B0 = (float*)d_ws;
    float* B1 = B0 + ED;
    float* B2 = B1 + ED;
    float* B3 = B2 + ED;
    float* B4 = B3 + ED;
    unsigned short* wb = (unsigned short*)(B4 + ED);

    const int SQ = 256 * 256;
    const int RQ = 64 * 256;
    unsigned short* t_W1   = wb;
    unsigned short* t_Wlgm = t_W1   + SQ;
    unsigned short* t_Wr1a = t_Wlgm + SQ;
    unsigned short* t_Wr1b = t_Wr1a + SQ;
    unsigned short* t_Wr2a = t_Wr1b + SQ;
    unsigned short* t_Wr2b = t_Wr2a + SQ;
    unsigned short* t_Wr3a = t_Wr2b + SQ;
    unsigned short* t_Wr3b = t_Wr3a + SQ;
    unsigned short* t_W2   = t_Wr3b + SQ;
    unsigned short* t_Wol0 = t_W2   + SQ;
    unsigned short* t_Wo4  = t_Wol0 + SQ;
    unsigned short* t_Wlge = t_Wo4  + SQ;
    unsigned short* t_Woe  = t_Wlge + RQ;

    {
        const int bS = (SQ + 255) / 256, bR = (RQ + 255) / 256;
        k_convert_transpose<<<bS, 256, 0, stream>>>(W1,   t_W1,   256);
        k_convert_transpose<<<bS, 256, 0, stream>>>(Wlgm, t_Wlgm, 256);
        k_convert_transpose<<<bS, 256, 0, stream>>>(Wr1a, t_Wr1a, 256);
        k_convert_transpose<<<bS, 256, 0, stream>>>(Wr1b, t_Wr1b, 256);
        k_convert_transpose<<<bS, 256, 0, stream>>>(Wr2a, t_Wr2a, 256);
        k_convert_transpose<<<bS, 256, 0, stream>>>(Wr2b, t_Wr2b, 256);
        k_convert_transpose<<<bS, 256, 0, stream>>>(Wr3a, t_Wr3a, 256);
        k_convert_transpose<<<bS, 256, 0, stream>>>(Wr3b, t_Wr3b, 256);
        k_convert_transpose<<<bS, 256, 0, stream>>>(W2,   t_W2,   256);
        k_convert_transpose<<<bS, 256, 0, stream>>>(Wol0, t_Wol0, 256);
        k_convert_transpose<<<bS, 256, 0, stream>>>(Wo4,  t_Wo4,  256);
        k_convert_transpose<<<bR, 256, 0, stream>>>(Wlge, t_Wlge, 64);
        k_convert_transpose<<<bR, 256, 0, stream>>>(Woe,  t_Woe,  64);
    }

    float* out_x4 = (float*)d_out;        // [E,256]
    float* out_x5 = out_x4 + ED;          // [N,256]

    // prev = relu(msg @ W1 + b1)                       -> B0
    run_gemm(msg, t_W1, b1, nullptr, B0, E, 256, 1, stream);
    // elg = lg_edge_attr @ Wlg_e                       -> B1..B3 (ELG rows)
    run_gemm(lgattr, t_Wlge, nullptr, nullptr, B1, ELG, 64, 0, stream);
    // agg = segment_sum(msg[src] * elg, dst)           -> B4
    {
        long long n4 = ED >> 2;
        k_zero_f32<<<dim3((unsigned)((n4 + 255) / 256)), 256, 0, stream>>>(B4, n4);
        long long tot = (long long)ELG * 64;
        k_lg_scatter<<<dim3((unsigned)((tot + 255) / 256)), 256, 0, stream>>>(
            msg, B1, lgidx, lgidx + ELG, B4, ELG);
    }
    // x0 = prev + relu(agg @ Wlg_m)                    -> B1 (elg dead)
    run_gemm(B4, t_Wlgm, nullptr, B0, B1, E, 256, 1, stream);
    // residual block 1
    run_gemm(B1, t_Wr1a, br1a, nullptr, B2, E, 256, 1, stream);
    run_gemm(B2, t_Wr1b, br1b, B1, B3, E, 256, 1, stream);
    // x1 = relu(x @ W2 + b2)                           -> B0 (prev dead)
    run_gemm(B3, t_W2, b2, nullptr, B0, E, 256, 1, stream);
    // residual block 2
    run_gemm(B0, t_Wr2a, br2a, nullptr, B1, E, 256, 1, stream);
    run_gemm(B1, t_Wr2b, br2b, B0, B2, E, 256, 1, stream);
    // residual block 3 -> x4 straight into d_out
    run_gemm(B2, t_Wr3a, br3a, nullptr, B3, E, 256, 1, stream);
    run_gemm(B3, t_Wr3b, br3b, B2, out_x4, E, 256, 1, stream);
    // ee = g_edge_attr @ Wo_e + bo_e                   -> B0
    run_gemm(gattr, t_Woe, boe, nullptr, B0, E, 64, 0, stream);
    // node = segment_sum(ee * x4, g_edge_index[1])     -> B1
    {
        long long n4 = ND >> 2;
        k_zero_f32<<<dim3((unsigned)((n4 + 255) / 256)), 256, 0, stream>>>(B1, n4);
        long long tot = (long long)E * 64;
        k_node_scatter<<<dim3((unsigned)((tot + 255) / 256)), 256, 0, stream>>>(
            B0, out_x4, gidx + E, B1, E);
    }
    // output MLP (Wo_l0 applied 3x), then x5 = h @ Wo_4 -> d_out tail
    run_gemm(B1, t_Wol0, bol0, nullptr, B2, N, 256, 1, stream);
    run_gemm(B2, t_Wol0, bol0, nullptr, B3, N, 256, 1, stream);
    run_gemm(B3, t_Wol0, bol0, nullptr, B4, N, 256, 1, stream);
    run_gemm(B4, t_Wo4, nullptr, nullptr, out_x5, N, 256, 0, stream);
}
